// FractalSSM_57672820851323
// MI455X (gfx1250) — compile-verified, matrix operand order checked by reading, not run on hardware
//
#include <hip/hip_runtime.h>
#include <hip/hip_bf16.h>

typedef __attribute__((ext_vector_type(16))) _Float16 v16h;
typedef __attribute__((ext_vector_type(8)))  _Float16 v8h;
typedef __attribute__((ext_vector_type(8)))  float    v8f;

union Frag16 { v16h v; struct { v8h lo, hi; } p; };

__device__ __forceinline__ float silu_f(float x)    { return x / (1.f + __expf(-x)); }
__device__ __forceinline__ float softplus_f(float x){ return x > 20.f ? x : log1pf(__expf(x)); }

#define BM 128
#define BN 128
#define BK 32

// Generic WMMA GEMM: out = epilogue(A @ B)
// Static contract (all launches obey): M % 128 == 0, K % 32 == 0. Only N may be ragged.
// A: row-major MxK (f32 or f16), B: row-major KxN f32 (converted to f16 in LDS).
// EPI 0: store f32 (and f16 shadow if out0h) ; ldc = N
// EPI 1: col<Dsplit -> silu -> out0(f32)+out0h(f16), else raw -> out1 ; split ldc
// EPI 2: softplus(acc + bias[col]) -> out0 ; ldc = N
// EPI 3: acc + bias[col] -> out0 ; ldc = N
template<int EPI, bool AHALF>
__global__ __launch_bounds__(256)
void wmma_gemm_k(const void* __restrict__ Ap, const float* __restrict__ Bw,
                 float* __restrict__ out0, _Float16* __restrict__ out0h,
                 float* __restrict__ out1, const float* __restrict__ bias,
                 int M, int N, int K, int lda, int ldb, int Dsplit)
{
  __shared__ __align__(16) _Float16 As[BM * BK];
  __shared__ __align__(16) _Float16 Bs[BN * BK];   // [col][k]
  const int tid   = threadIdx.x;
  const int wm    = tid >> 5;      // wave id 0..7 -> M strip
  const int lane  = tid & 31;
  const int halfq = lane >> 4;     // 0 / 1 half of wave
  const int lidx  = lane & 15;
  const long blockM = (long)blockIdx.y * BM;
  const long blockN = (long)blockIdx.x * BN;

  v8f acc[8];
#pragma unroll
  for (int i = 0; i < 8; ++i)
#pragma unroll
    for (int j = 0; j < 8; ++j) acc[i][j] = 0.f;

  // staging thread mapping
  const int ar   = tid >> 1;           // A: row 0..127
  const int akp  = (tid & 1) * 16;     // A: k half 0 / 16
  const int bcol = tid & 127;          // B: col 0..127
  const int bkp  = (tid >> 7) * 16;    // B: k half 0 / 16
  const long arow = blockM + ar;       // always < M (M % 128 == 0)
  const long gcolRaw = blockN + bcol;
  const long gcolB = gcolRaw < N ? gcolRaw : (long)(N - 1);  // branch-free ragged N
  const float cmask = gcolRaw < N ? 1.f : 0.f;

  // double-buffer staging registers
  uint4  aq[2];    // AHALF path: 16 halves
  float4 af[4];    // f32 path: 16 floats
  float  bf[16];

  auto loadTiles = [&](int kbase) {
    if (AHALF) {
      const _Float16* Ag = (const _Float16*)Ap;
      const uint4* p = (const uint4*)(Ag + arow * (long)lda + kbase + akp);
      aq[0] = p[0]; aq[1] = p[1];
    } else {
      const float* Ag = (const float*)Ap;
      const float4* p = (const float4*)(Ag + arow * (long)lda + kbase + akp);
      af[0] = p[0]; af[1] = p[1]; af[2] = p[2]; af[3] = p[3];
    }
    const float* bp = Bw + (long)(kbase + bkp) * ldb + gcolB;
#pragma unroll
    for (int i = 0; i < 16; ++i) bf[i] = bp[(long)i * ldb];
  };

  auto storeTiles = [&]() {
    _Float16* adst = &As[ar * BK + akp];
    if (AHALF) {
      *(uint4*)adst       = aq[0];
      *(uint4*)(adst + 8) = aq[1];
    } else {
      const float* afp = (const float*)af;
      v8h h0, h1;
#pragma unroll
      for (int i = 0; i < 8; ++i) { h0[i] = (_Float16)afp[i]; h1[i] = (_Float16)afp[8 + i]; }
      *(v8h*)adst       = h0;
      *(v8h*)(adst + 8) = h1;
    }
    _Float16* bdst = &Bs[bcol * BK + bkp];
    v8h g0, g1;
#pragma unroll
    for (int i = 0; i < 8; ++i) { g0[i] = (_Float16)(bf[i] * cmask); g1[i] = (_Float16)(bf[8 + i] * cmask); }
    *(v8h*)bdst       = g0;
    *(v8h*)(bdst + 8) = g1;
  };

  loadTiles(0);
  for (int k0 = 0; k0 < K; k0 += BK) {
    storeTiles();
    __syncthreads();
    if (k0 + BK < K) {
      loadTiles(k0 + BK);   // prefetch next chunk into VGPRs while WMMAs run
      __builtin_prefetch(Bw + (long)(k0 + 2 * BK) * ldb + gcolB, 0, 1);
    }

    // A fragment per ISA 16-bit 16x32 layout: lanes 0-15 K{0..7,16..23}, lanes 16-31 K{8..15,24..31}
    Frag16 afrag;
    const _Float16* ab = &As[(wm * 16 + lidx) * BK + halfq * 8];
    afrag.p.lo = *(const v8h*)(ab);
    afrag.p.hi = *(const v8h*)(ab + 16);
#pragma unroll
    for (int nt = 0; nt < 8; ++nt) {
      Frag16 bfrag;
      const _Float16* bb = &Bs[(nt * 16 + lidx) * BK + halfq * 8];
      bfrag.p.lo = *(const v8h*)(bb);
      bfrag.p.hi = *(const v8h*)(bb + 16);
      acc[nt] = __builtin_amdgcn_wmma_f32_16x16x32_f16(
          false, afrag.v, false, bfrag.v, (short)0, acc[nt], false, false);
    }
    __syncthreads();
  }

  // Epilogue. C/D layout: VGPR r, lanes 0-15 -> M=r, lanes 16-31 -> M=r+8, N=lane&15.
#pragma unroll
  for (int nt = 0; nt < 8; ++nt) {
    const long gcol = blockN + nt * 16 + lidx;
    if (gcol >= N) continue;
#pragma unroll
    for (int r = 0; r < 8; ++r) {
      const long grow = blockM + wm * 16 + r + halfq * 8;
      float v = acc[nt][r];
      if (EPI == 0) {
        out0[grow * (long)N + gcol] = v;
        if (out0h) out0h[grow * (long)N + gcol] = (_Float16)v;
      } else if (EPI == 1) {
        if (gcol < Dsplit) {
          const float s = silu_f(v);
          out0[grow * (long)Dsplit + gcol]  = s;
          out0h[grow * (long)Dsplit + gcol] = (_Float16)s;
        } else {
          out1[grow * (long)(N - Dsplit) + (gcol - Dsplit)] = v;
        }
      } else if (EPI == 2) {
        out0[grow * (long)N + gcol] = softplus_f(v + bias[gcol]);
      } else {
        out0[grow * (long)N + gcol] = v + bias[gcol];
      }
    }
  }
}

#define SS 2048
#define DD 1024
#define NN 16

// Sequential fractal scan. One workgroup per batch (1024 threads = 32 wave32),
// one thread per d; state h[d, n, c] lives in 32 VGPRs per thread.
// Per-step broadcast staging uses CDNA5 async global->LDS loads (ASYNCcnt);
// cross-D h_carrier reduction: wave shfl_xor butterfly + ds_add_f32 LDS atomics.
__global__ __launch_bounds__(1024)
void scan_k(const float* __restrict__ xin, const float* __restrict__ dtp,
            const float* __restrict__ xpp, const float* __restrict__ xresp,
            const float* __restrict__ zp, const float* __restrict__ alog,
            float* __restrict__ ys)
{
  const int b = blockIdx.x;
  const int d = threadIdx.x;
  __shared__ float hsum[2 * NN];
  __shared__ float Tl[NN];
  __shared__ float bcs[4 * NN];   // [0..31]=B_t, [32..63]=C_t
  __shared__ float xrs[2 * NN];
  __shared__ float Aneg[NN];
  if (d < NN) Aneg[d] = -__expf(alog[d]);

  float hr[NN], hi[NN];
#pragma unroll
  for (int n = 0; n < NN; ++n) {   // linspace(0, 2pi(1-1/N), N) -> 2*pi*n/N
    const float ph = 6.2831853071795864f * (float)n / (float)NN;
    hr[n] = 0.01f * __cosf(ph);
    hi[n] = 0.01f * __sinf(ph);
  }

  for (int t = 0; t < SS; ++t) {
    const long row = (long)b * SS + t;
    __syncthreads();                       // previous step done reading LDS
    if (d >= 32 && d < 96) {               // B_t / C_t : 64 floats
      const unsigned long long g = (unsigned long long)(const void*)(xpp + row * 128 + 64 + (d - 32));
      const unsigned int l = (unsigned int)(unsigned long long)(const void*)&bcs[d - 32];
      asm volatile("global_load_async_to_lds_b32 %0, %1, off" :: "v"(l), "v"(g) : "memory");
    } else if (d >= 96 && d < 128) {       // xres_t : 32 floats
      const unsigned long long g = (unsigned long long)(const void*)(xresp + row * (2 * NN) + (d - 96));
      const unsigned int l = (unsigned int)(unsigned long long)(const void*)&xrs[d - 96];
      asm volatile("global_load_async_to_lds_b32 %0, %1, off" :: "v"(l), "v"(g) : "memory");
    }
    if (d < 32) hsum[d] = 0.f;
    asm volatile("s_wait_asynccnt 0x0" ::: "memory");
    __syncthreads();

    // h_carrier: sum over d for each of 32 channels
#pragma unroll
    for (int ch = 0; ch < 2 * NN; ++ch) {
      float v = (ch & 1) ? hi[ch >> 1] : hr[ch >> 1];
#pragma unroll
      for (int off = 16; off >= 1; off >>= 1) v += __shfl_xor(v, off, 32);
      if ((d & 31) == 0) atomicAdd(&hsum[ch], v);
    }
    __syncthreads();
    if (d < NN) {      // phase of mean == phase of sum (positive scale)
      const float phc = atan2f(hsum[d * 2 + 1], hsum[d * 2] + 1e-10f);
      const float phx = atan2f(xrs[d * 2 + 1], xrs[d * 2] + 1e-10f);
      const float cc  = __cosf((phx - phc) * 0.5f);
      Tl[d] = cc * cc;
    }
    __syncthreads();

    const long idx  = row * DD + d;
    const float x_t = xin[idx];
    const float dt_t = dtp[idx];
    const float dx  = dt_t * x_t;
    float y = 0.f;
#pragma unroll
    for (int n = 0; n < NN; ++n) {
      const float alpha = __expf(dt_t * Aneg[n]);
      const float expo  = 1.f + (1.f - alpha) * 0.5f * (1.f - Tl[n]);
      float ar = hr[n], ai = hi[n];
      const float mag  = __fsqrt_rn(ar * ar + ai * ai + 1e-8f);
      const float ph   = atan2f(ai, ar + 1e-10f);
      const float cmag = fminf(__expf(expo * __logf(mag + 1e-8f)), 10.f);
      ar = cmag * __cosf(ph);
      ai = cmag * __sinf(ph);
      ar += bcs[n * 2]     * dx;
      ai += bcs[n * 2 + 1] * dx;
      const float m2  = __fsqrt_rn(ar * ar + ai * ai + 1e-8f);
      const float ph2 = atan2f(ai, ar + 1e-10f);
      const float tm  = tanhf(m2);
      ar = tm * __cosf(ph2);
      ai = tm * __sinf(ph2);
      hr[n] = ar; hi[n] = ai;
      y += bcs[2 * NN + n * 2] * ar + bcs[2 * NN + n * 2 + 1] * ai;
    }
    const float zv = zp[idx];
    ys[idx] = y * (zv / (1.f + __expf(-zv)));   // fuse y * silu(z)
  }
}

extern "C" void kernel_launch(void* const* d_in, const int* in_sizes, int n_in,
                              void* d_out, int out_size, void* d_ws, size_t ws_size,
                              hipStream_t stream)
{
  (void)in_sizes; (void)n_in; (void)out_size; (void)ws_size;
  const float* x      = (const float*)d_in[0];
  const float* W_in   = (const float*)d_in[1];
  const float* W_xprj = (const float*)d_in[2];
  const float* W_dt   = (const float*)d_in[3];
  const float* b_dt   = (const float*)d_in[4];
  const float* W_out  = (const float*)d_in[5];
  const float* b_out  = (const float*)d_in[6];
  const float* W_res  = (const float*)d_in[7];
  const float* A_log  = (const float*)d_in[8];

  const long Brows = 4L * 2048L;  // B*S = 8192

  char* w = (char*)d_ws;
  auto carve = [&](size_t bytes) { char* p = w; w += (bytes + 255) & ~(size_t)255; return p; };
  float*    xin   = (float*)   carve(Brows * 1024 * 4);
  _Float16* xinh  = (_Float16*)carve(Brows * 1024 * 2);
  float*    z     = (float*)   carve(Brows * 1024 * 4);
  float*    xp    = (float*)   carve(Brows * 128 * 4);
  _Float16* xph   = (_Float16*)carve(Brows * 128 * 2);
  float*    dt    = (float*)   carve(Brows * 1024 * 4);
  float*    xres  = (float*)   carve(Brows * 32 * 4);
  float*    ysbuf = (float*)   carve(Brows * 1024 * 4);

  dim3 blk(256);
  // 1) xz = x @ W_in ; x_in = silu(xz[:, :D]) (f32 + f16 shadow), z = xz[:, D:]
  wmma_gemm_k<1, false><<<dim3(2048 / BN, Brows / BM), blk, 0, stream>>>(
      x, W_in, xin, xinh, z, nullptr, (int)Brows, 2048, 1024, 1024, 2048, 1024);
  // 2) xp = x_in @ W_xproj (f32 + f16 shadow for dt GEMM)
  wmma_gemm_k<0, true><<<dim3(1, Brows / BM), blk, 0, stream>>>(
      xinh, W_xprj, xp, xph, nullptr, nullptr, (int)Brows, 128, 1024, 1024, 128, 0);
  // 3) dt = softplus(xp[:, :64] @ W_dt + b_dt)   (K=64, lda=128)
  wmma_gemm_k<2, true><<<dim3(1024 / BN, Brows / BM), blk, 0, stream>>>(
      xph, W_dt, dt, nullptr, nullptr, b_dt, (int)Brows, 1024, 64, 128, 1024, 0);
  // 4) xres = x_in @ W_res (N=32, ragged: column-clamped staging)
  wmma_gemm_k<0, true><<<dim3(1, Brows / BM), blk, 0, stream>>>(
      xinh, W_res, xres, nullptr, nullptr, nullptr, (int)Brows, 32, 1024, 1024, 32, 0);
  // 5) sequential fractal scan (fuses y * silu(z))
  scan_k<<<dim3(4), dim3(1024), 0, stream>>>(xin, dt, xp, xres, z, A_log, ysbuf);
  // 6) out = ys @ W_out + b_out
  wmma_gemm_k<3, false><<<dim3(1024 / BN, Brows / BM), blk, 0, stream>>>(
      ysbuf, W_out, (float*)d_out, nullptr, nullptr, b_out, (int)Brows, 1024, 1024, 1024, 1024, 0);
}
